// NaiveLSTM_47502338294019
// MI455X (gfx1250) — matrix operand order repaired and checked by reading, not compiled, not run on hardware
//
#include <hip/hip_runtime.h>
#include <math.h>

#define B_ 64
#define T_ 512
#define I_ 256
#define H_ 1024
#define K_TOT (I_ + H_)     // 1280: fused [x | h] contraction
#define SPLIT 4
#define KSPLIT (K_TOT / SPLIT)   // 320 per split wave

typedef __attribute__((ext_vector_type(16))) __bf16 v16bf;
typedef __attribute__((ext_vector_type(8)))  __bf16 v8bf;
typedef __attribute__((ext_vector_type(8)))  float  v8f;

// ---- workspace layout (bytes, all offsets 256B aligned) ----
#define OFF_W    ((size_t)0)              // 4*H*K_TOT bf16 = 10,485,760
#define OFF_X    ((size_t)10485760)       // T*B*I bf16     = 16,777,216
#define OFF_BIAS ((size_t)27262976)       // 4H f32         =     16,384
#define OFF_A0   ((size_t)27279360)       // B*K_TOT bf16   =    163,840
#define OFF_A1   ((size_t)27443200)       // B*K_TOT bf16   =    163,840
#define OFF_C    ((size_t)27607040)       // B*H f32        =    262,144

// Fused weights W[g][n][k]: k<I -> w_i*[n,k], k>=I -> w_h*[n,k-I].
// Gate order matches reference: 0=i, 1=f, 2=g, 3=o.
__global__ void lstm_prep_w(const float* wi0, const float* wh0,
                            const float* wi1, const float* wh1,
                            const float* wi2, const float* wh2,
                            const float* wi3, const float* wh3,
                            __bf16* __restrict__ W) {
  size_t idx = (size_t)blockIdx.x * blockDim.x + threadIdx.x;
  const size_t N = (size_t)4 * H_ * K_TOT;
  if (idx >= N) return;
  int g = (int)(idx / ((size_t)H_ * K_TOT));
  size_t r = idx % ((size_t)H_ * K_TOT);
  int n = (int)(r / K_TOT);
  int k = (int)(r % K_TOT);
  const float* wi = (g == 0) ? wi0 : (g == 1) ? wi1 : (g == 2) ? wi2 : wi3;
  const float* wh = (g == 0) ? wh0 : (g == 1) ? wh1 : (g == 2) ? wh2 : wh3;
  float v = (k < I_) ? wi[(size_t)n * I_ + k] : wh[(size_t)n * H_ + (k - I_)];
  W[idx] = (__bf16)v;
}

// inputs (B,T,I) f32 -> Xbf (T,B,I) bf16 (per-step slices become contiguous)
__global__ void lstm_prep_x(const float* __restrict__ x, __bf16* __restrict__ xbf) {
  size_t idx = (size_t)blockIdx.x * blockDim.x + threadIdx.x;
  const size_t N = (size_t)T_ * B_ * I_;
  if (idx >= N) return;
  int t = (int)(idx / (B_ * I_));
  int r = (int)(idx % (B_ * I_));
  int b = r / I_;
  int i = r % I_;
  xbf[idx] = (__bf16)x[(size_t)b * T_ * I_ + (size_t)t * I_ + i];
}

// A0 = [x_0 | h_0] bf16, c = c0, combined bias
__global__ void lstm_prep_state(const float* __restrict__ inputs, const float* __restrict__ h0,
                                const float* __restrict__ c0,
                                const float* bi0, const float* bh0,
                                const float* bi1, const float* bh1,
                                const float* bi2, const float* bh2,
                                const float* bi3, const float* bh3,
                                __bf16* __restrict__ A0, float* __restrict__ c,
                                float* __restrict__ bias) {
  int idx = blockIdx.x * blockDim.x + threadIdx.x;   // 0 .. B*K_TOT-1
  if (idx >= B_ * K_TOT) return;
  int b = idx / K_TOT;
  int k = idx % K_TOT;
  float v = (k < I_) ? inputs[(size_t)b * T_ * I_ + k]          // x at t=0
                     : h0[(size_t)b * H_ + (k - I_)];
  A0[idx] = (__bf16)v;
  if (idx < B_ * H_) c[idx] = c0[idx];
  if (idx < 4 * H_) {
    int g = idx / H_, n = idx % H_;
    const float* bi = (g == 0) ? bi0 : (g == 1) ? bi1 : (g == 2) ? bi2 : bi3;
    const float* bh = (g == 0) ? bh0 : (g == 1) ? bh1 : (g == 2) ? bh2 : bh3;
    bias[idx] = bi[n] + bh[n];
  }
}

__device__ __forceinline__ v16bf ldfrag(const __bf16* p, int delta) {
  v8bf lo = *(const v8bf*)p;
  v8bf hi = *(const v8bf*)(p + delta);
  v16bf r;
#pragma unroll
  for (int e = 0; e < 8; ++e) { r[e] = lo[e]; r[e + 8] = hi[e]; }
  return r;
}

// One LSTM step, split-K x4 per 16x16 tile:
//   - block = 256 threads = 8 waves = 2 tiles x 4 K-split waves
//   - each split wave runs 10 K-chunks x 4 gates of v_wmma_f32_16x16x32_bf16
//   - f32 partials reduced through LDS, epilogue split across the 4 waves
// 1024 waves/step total -> ~2 waves per SIMD chip-wide for latency hiding.
__global__ __launch_bounds__(256)
void lstm_step(const __bf16* __restrict__ Acur,   // (B, K_TOT): [x_t | h_t]
               __bf16* __restrict__ Anext,        // (B, K_TOT): gets h_{t+1} (+ x_{t+1})
               const __bf16* __restrict__ xnext,  // (B, I) slice t+1, or null at t=T-1
               float* __restrict__ c,             // (B, H)
               const __bf16* __restrict__ W,      // (4, H, K_TOT)
               const float* __restrict__ bias,    // (4H)
               float* __restrict__ hseq,          // (B, T, H)
               int t) {
  __shared__ float red[2 * SPLIT * 4 * 256];      // [tib][split][gate][v*32+lane], 32KB

  const int wave = threadIdx.x >> 5;
  const int lane = threadIdx.x & 31;
  const int half = lane >> 4;
  const int lr   = lane & 15;
  const int tib  = wave >> 2;               // tile in block: 0..1
  const int s    = wave & 3;                // K-split id:    0..3
  const int tile = blockIdx.x * 2 + tib;    // 0..255
  const int rt = tile >> 6;                 // 0..3  (B/16)
  const int ct = tile & 63;                 // 0..63 (H/16)

  const int arow = rt * 16 + lr;            // A: all lanes map M = lane%16
  const int ncol = ct * 16 + lr;            // B/C/D column for this lane

  // Single base pointers; every K-iteration uses pure immediate offsets.
  const __bf16* ap = Acur + (size_t)arow * K_TOT + s * KSPLIT + half * 16;
  const int wo = s * KSPLIT + half * 16;
  const __bf16* w0 = W + ((size_t)0 * H_ + ncol) * K_TOT + wo;
  const __bf16* w1 = W + ((size_t)1 * H_ + ncol) * K_TOT + wo;
  const __bf16* w2 = W + ((size_t)2 * H_ + ncol) * K_TOT + wo;
  const __bf16* w3 = W + ((size_t)3 * H_ + ncol) * K_TOT + wo;

  v8f acc0 = {}, acc1 = {}, acc2 = {}, acc3 = {};

  // 2-stage software pipeline: loads for kb+32 issue before the WMMAs for kb.
  v16bf a  = ldfrag(ap, 8);
  v16bf f0 = ldfrag(w0, 8);
  v16bf f1 = ldfrag(w1, 8);
  v16bf f2 = ldfrag(w2, 8);
  v16bf f3 = ldfrag(w3, 8);
#pragma unroll
  for (int kb = 0; kb < KSPLIT; kb += 32) {
    v16bf an = a, g0 = f0, g1 = f1, g2 = f2, g3 = f3;
    if (kb + 32 < KSPLIT) {
      an = ldfrag(ap + kb + 32, 8);
      g0 = ldfrag(w0 + kb + 32, 8);
      g1 = ldfrag(w1 + kb + 32, 8);
      g2 = ldfrag(w2 + kb + 32, 8);
      g3 = ldfrag(w3 + kb + 32, 8);
    }
    acc0 = __builtin_amdgcn_wmma_f32_16x16x32_bf16(false, a, false, f0, (short)0, acc0, false, false);
    acc1 = __builtin_amdgcn_wmma_f32_16x16x32_bf16(false, a, false, f1, (short)0, acc1, false, false);
    acc2 = __builtin_amdgcn_wmma_f32_16x16x32_bf16(false, a, false, f2, (short)0, acc2, false, false);
    acc3 = __builtin_amdgcn_wmma_f32_16x16x32_bf16(false, a, false, f3, (short)0, acc3, false, false);
    a = an; f0 = g0; f1 = g1; f2 = g2; f3 = g3;
  }

  // Dump partial sums to LDS: red[tib][s][g][v*32 + lane].
  {
    float* base = red + (size_t)(tib * SPLIT + s) * 4 * 256;
#pragma unroll
    for (int v = 0; v < 8; ++v) {
      base[0 * 256 + v * 32 + lane] = acc0[v];
      base[1 * 256 + v * 32 + lane] = acc1[v];
      base[2 * 256 + v * 32 + lane] = acc2[v];
      base[3 * 256 + v * 32 + lane] = acc3[v];
    }
  }
  __syncthreads();

  // Reduction + fused cell update, split across the 4 waves:
  // split wave s handles accumulator rows v = {2s, 2s+1}.
  // (v, lane) -> (row = rt*16 + half*8 + v, col = ncol) — same mapping as the
  // producer, so reading index v*32+lane with our own lane is consistent.
  const float bi  = bias[ncol];
  const float bff = bias[H_ + ncol];
  const float bg  = bias[2 * H_ + ncol];
  const float bo  = bias[3 * H_ + ncol];
  const float* rb = red + (size_t)tib * SPLIT * 4 * 256;
#pragma unroll
  for (int vv = 0; vv < 2; ++vv) {
    const int v = s * 2 + vv;
    float zi = bi, zf = bff, zg = bg, zo = bo;
#pragma unroll
    for (int ss = 0; ss < SPLIT; ++ss) {
      const float* p = rb + (size_t)ss * 4 * 256 + v * 32 + lane;
      zi += p[0 * 256];
      zf += p[1 * 256];
      zg += p[2 * 256];
      zo += p[3 * 256];
    }
    float ig = 1.0f / (1.0f + __expf(-zi));
    float fg = 1.0f / (1.0f + __expf(-zf));
    float gg = tanhf(zg);
    float og = 1.0f / (1.0f + __expf(-zo));
    const int row = rt * 16 + half * 8 + v;
    const size_t sidx = (size_t)row * H_ + ncol;
    float cn = fg * c[sidx] + ig * gg;
    c[sidx] = cn;
    float hn = og * tanhf(cn);
    hseq[(size_t)row * (T_ * H_) + (size_t)t * H_ + ncol] = hn;
    Anext[(size_t)row * K_TOT + I_ + ncol] = (__bf16)hn;
  }

  // Stage x_{t+1} into the next A buffer (vectorized: 2048 uint4 copies).
  if (xnext != nullptr) {
    const int tid = blockIdx.x * blockDim.x + threadIdx.x;
    if (tid < (B_ * I_ * 2) / 16) {                          // 2048
      const int b = tid >> 5;         // 32 uint4 per row (I*2/16)
      const int ch = tid & 31;
      const uint4* src = (const uint4*)(xnext) + tid;        // (B,I) contiguous
      uint4* dst = (uint4*)(Anext + (size_t)b * K_TOT) + ch;
      *dst = *src;
    }
  }
}

// final h = hseq[:, T-1, :], final c = c state
__global__ void lstm_final(const float* __restrict__ c, float* __restrict__ out) {
  int idx = blockIdx.x * blockDim.x + threadIdx.x;
  if (idx >= B_ * H_) return;
  int b = idx / H_, n = idx % H_;
  const size_t BTH = (size_t)B_ * T_ * H_;
  out[BTH + idx] = out[(size_t)b * (T_ * H_) + (size_t)(T_ - 1) * H_ + n];
  out[BTH + (size_t)B_ * H_ + idx] = c[idx];
}

extern "C" void kernel_launch(void* const* d_in, const int* in_sizes, int n_in,
                              void* d_out, int out_size, void* d_ws, size_t ws_size,
                              hipStream_t stream) {
  const float* inputs = (const float*)d_in[0];
  const float* h0     = (const float*)d_in[1];
  const float* c0     = (const float*)d_in[2];
  const float* w_ii = (const float*)d_in[3];
  const float* w_hi = (const float*)d_in[4];
  const float* b_ii = (const float*)d_in[5];
  const float* b_hi = (const float*)d_in[6];
  const float* w_if = (const float*)d_in[7];
  const float* w_hf = (const float*)d_in[8];
  const float* b_if = (const float*)d_in[9];
  const float* b_hf = (const float*)d_in[10];
  const float* w_io = (const float*)d_in[11];
  const float* w_ho = (const float*)d_in[12];
  const float* b_io = (const float*)d_in[13];
  const float* b_ho = (const float*)d_in[14];
  const float* w_ig = (const float*)d_in[15];
  const float* w_hg = (const float*)d_in[16];
  const float* b_ig = (const float*)d_in[17];
  const float* b_hg = (const float*)d_in[18];

  char* ws = (char*)d_ws;
  __bf16* Wbf  = (__bf16*)(ws + OFF_W);
  __bf16* Xbf  = (__bf16*)(ws + OFF_X);
  float*  bias = (float*)(ws + OFF_BIAS);
  __bf16* A0   = (__bf16*)(ws + OFF_A0);
  __bf16* A1   = (__bf16*)(ws + OFF_A1);
  float*  c    = (float*)(ws + OFF_C);
  float*  out  = (float*)d_out;

  {
    size_t N = (size_t)4 * H_ * K_TOT;
    lstm_prep_w<<<(unsigned)((N + 255) / 256), 256, 0, stream>>>(
        w_ii, w_hi, w_if, w_hf, w_ig, w_hg, w_io, w_ho, Wbf);
  }
  {
    size_t N = (size_t)T_ * B_ * I_;
    lstm_prep_x<<<(unsigned)((N + 255) / 256), 256, 0, stream>>>(inputs, Xbf);
  }
  lstm_prep_state<<<(B_ * K_TOT + 255) / 256, 256, 0, stream>>>(
      inputs, h0, c0, b_ii, b_hi, b_if, b_hf, b_ig, b_hg, b_io, b_ho, A0, c, bias);

  // sequential scan: 512 stream-ordered step kernels (graph capture amortizes launch cost)
  for (int t = 0; t < T_; ++t) {
    __bf16* Acur  = (t & 1) ? A1 : A0;
    __bf16* Anext = (t & 1) ? A0 : A1;
    const __bf16* xnext = (t + 1 < T_) ? (Xbf + (size_t)(t + 1) * B_ * I_) : nullptr;
    lstm_step<<<128, 256, 0, stream>>>(Acur, Anext, xnext, c, Wbf, bias, out, t);
  }

  lstm_final<<<(B_ * H_ + 255) / 256, 256, 0, stream>>>(c, out);
}
